// GraphCare_27960237097327
// MI455X (gfx1250) — matrix operand order repaired and checked by reading, not compiled, not run on hardware
//
#include <hip/hip_runtime.h>

#define BSZ   32
#define NVIS  6
#define MF    12
#define MT    16
#define NF    1992
#define NR    11
#define HID   128
#define RED   256
#define OUTD  90

#define ROWS   (BSZ*NVIS*MF)   // 2304
#define MTILES (ROWS/16)       // 144
#define KPAD   2016            // NF padded to multiple of 32
#define KCN    (KPAD/32)       // 63 k-chunks
#define NTN    (RED/16)        // 16 n-tiles

typedef __attribute__((ext_vector_type(16))) __bf16 v16bf;
typedef __attribute__((ext_vector_type(8)))  float  v8f;

union ABu { uint4 q[2]; v16bf v; unsigned short s[16]; };

__device__ __forceinline__ unsigned short f32_bf16(float f) {
    unsigned int u = __float_as_uint(f);
    u += 0x7fffu + ((u >> 16) & 1u);            // round-to-nearest-even
    return (unsigned short)(u >> 16);
}

// ---------------------------------------------------------------------------
// K0: reduced_emb = emb @ W_v^T + b_v  -> bf16 in WMMA-B swizzled layout
//     reduced_edge = W_edge @ W_r^T + b_r -> f32
// blocks 0..KPAD-1 handle one K-row of B; blocks KPAD..KPAD+NR-1 handle edges.
// ---------------------------------------------------------------------------
__global__ void k0_prep(const float* __restrict__ emb, const float* __restrict__ Wv,
                        const float* __restrict__ bv,  const float* __restrict__ Wedge,
                        const float* __restrict__ Wr,  const float* __restrict__ br,
                        unsigned short* __restrict__ Bswz, float* __restrict__ redge) {
    __shared__ float srow[HID];
    int blk = blockIdx.x;
    int r = threadIdx.x;                        // 0..255 (output column)
    if (blk < KPAD) {
        int n = blk;
        float val = 0.f;
        if (n < NF) {
            if (threadIdx.x < HID) srow[threadIdx.x] = emb[(size_t)n*HID + threadIdx.x];
            __syncthreads();
            float acc = bv[r];
            for (int j = 0; j < HID; ++j) acc += srow[j] * Wv[(size_t)r*HID + j];
            val = acc;
        } else {
            __syncthreads();                    // keep barriers uniform
        }
        // B layout: lane L<16 holds K=0..15 (slot=K) for N=L; L>=16 holds K=16..31
        int kc = n >> 5, kr = n & 31;
        int nt = r >> 4;
        int lane = (r & 15) + ((kr >= 16) ? 16 : 0);
        int slot = kr & 15;
        Bswz[(((size_t)kc*NTN + nt)*32 + lane)*16 + slot] = f32_bf16(val);
    } else {
        int rr = blk - KPAD;                    // 0..10
        if (threadIdx.x < HID) srow[threadIdx.x] = Wedge[(size_t)rr*HID + threadIdx.x];
        __syncthreads();
        float acc = br[r];
        for (int j = 0; j < HID; ++j) acc += srow[j] * Wr[(size_t)r*HID + j];
        redge[rr*RED + r] = acc;
    }
}

// ---------------------------------------------------------------------------
// K1: alpha/softmax/beta -> attn[b,v,m] = alpha*beta, and featsum[b,v,m]
// One block per (b,v); 4 waves, each wave handles 3 m-rows (dot over 1992).
// ---------------------------------------------------------------------------
__global__ void k1_attn(const float* __restrict__ FI, const float* __restrict__ Walpha,
                        const float* __restrict__ balpha, const float* __restrict__ Wbeta,
                        const float* __restrict__ bbeta, float* __restrict__ attn,
                        float* __restrict__ fsum) {
    __shared__ float sdot[MF], sfs[MF];
    int bv = blockIdx.x;                        // 0..191
    int v = bv % NVIS;
    int lane = threadIdx.x & 31;
    int wave = threadIdx.x >> 5;                // 0..3
    for (int m = wave; m < MF; m += 4) {
        const float* row = FI + ((size_t)bv*MF + m)*NF;
        float d = 0.f, f = 0.f;
        for (int n = lane; n < NF; n += 32) {
            float x = row[n];
            d += Walpha[n]*x;
            f += x;
        }
        for (int off = 16; off > 0; off >>= 1) {
            d += __shfl_xor(d, off, 32);
            f += __shfl_xor(f, off, 32);
        }
        if (lane == 0) { sdot[m] = d; sfs[m] = f; }
    }
    __syncthreads();
    if (threadIdx.x == 0) {
        float e[MF];
        float mx = -1e30f;
        for (int m = 0; m < MF; ++m) { e[m] = sdot[m] + balpha[m]; mx = fmaxf(mx, e[m]); }
        float s = 0.f;
        for (int m = 0; m < MF; ++m) { e[m] = expf(e[m] - mx); s += e[m]; }
        float braw = bbeta[v];
        for (int m = 0; m < MF; ++m) { e[m] /= s; braw += Wbeta[m]*e[m]; }
        float lamb = expf(-0.1f * (float)(NVIS - v));
        float beta = tanhf(braw) * lamb;
        for (int m = 0; m < MF; ++m) {
            attn[bv*MF + m] = e[m]*beta;
            fsum[bv*MF + m] = sfs[m];
        }
    }
}

// ---------------------------------------------------------------------------
// K2a: t-reduce neighbor_index -> bf16 A tiles in WMMA-A swizzled layout.
// One block per (mtile, kc); streams the 293 MB tensor fully coalesced.
// ---------------------------------------------------------------------------
__global__ void k2a_tred(const float* __restrict__ NI, unsigned short* __restrict__ Aswz) {
    __shared__ float tile[16][32];
    int gid = blockIdx.x;
    int mtile = gid / KCN;
    int kc = gid % KCN;
    int tid = threadIdx.x;                      // 256
    int kk = kc*32 + (tid & 31);
    int r0 = tid >> 5;                          // 0..7
    float s0 = 0.f, s1 = 0.f;
    if (kk < NF) {
        size_t b0 = ((size_t)(mtile*16 + r0)    ) * MT * NF + kk;
        size_t b1 = ((size_t)(mtile*16 + r0 + 8)) * MT * NF + kk;
        for (int t = 0; t < MT; ++t) {
            s0 += NI[b0 + (size_t)t*NF];
            s1 += NI[b1 + (size_t)t*NF];
        }
    }
    tile[r0    ][tid & 31] = s0;
    tile[r0 + 8][tid & 31] = s1;
    __syncthreads();
    if (tid < 32) {
        // A layout: lanes 0-15: M=L, K=0..15; lanes 16-31: M=L-16, K=16..31
        int M = tid & 15, hi = tid >> 4;
        ABu u;
        for (int j = 0; j < 16; ++j) {
            int K = hi*16 + j;
            u.s[j] = f32_bf16(tile[M][K]);
        }
        uint4* outp = (uint4*)(Aswz + (((size_t)mtile*KCN + kc)*32 + tid)*16);
        outp[0] = u.q[0];
        outp[1] = u.q[1];
    }
}

// ---------------------------------------------------------------------------
// K2b: bf16 WMMA GEMM (16 rows x 256 cols per block) + fused epilogue:
//   h = relu(diag(W_l)*(attn*node_gemm + edge_agg) + b_l)
// 4 waves per block; wave w owns n-tiles 4w..4w+3.
// ---------------------------------------------------------------------------
__global__ void k2b_gemm(const unsigned short* __restrict__ Aswz,
                         const unsigned short* __restrict__ Bswz,
                         const float* __restrict__ relidx,
                         const float* __restrict__ redge,
                         const float* __restrict__ attn,
                         const float* __restrict__ Wl,
                         const float* __restrict__ bl,
                         float* __restrict__ h) {
    __shared__ float s_redge[NR*RED];           // 11 KB
    __shared__ float s_rels[16*NR];
    __shared__ float s_attn[16];
    int mtile = blockIdx.x;
    int lane = threadIdx.x & 31;
    int wave = threadIdx.x >> 5;

    for (int idx = threadIdx.x; idx < NR*RED; idx += blockDim.x) s_redge[idx] = redge[idx];
    for (int idx = threadIdx.x; idx < 16*NR; idx += blockDim.x) {
        int rl = idx / NR, rr = idx % NR;
        size_t base = ((size_t)(mtile*16 + rl)) * MT * NR + rr;
        float s = 0.f;
        for (int t = 0; t < MT; ++t) s += relidx[base + (size_t)t*NR];
        s_rels[idx] = s;
    }
    if (threadIdx.x < 16) s_attn[threadIdx.x] = attn[mtile*16 + threadIdx.x];
    __syncthreads();

    v8f zero = {0.f,0.f,0.f,0.f,0.f,0.f,0.f,0.f};
    v8f acc[4] = {zero, zero, zero, zero};

    for (int kc = 0; kc < KCN; ++kc) {
        ABu a;
        const uint4* ap = (const uint4*)(Aswz + (((size_t)mtile*KCN + kc)*32 + lane)*16);
        a.q[0] = ap[0];
        a.q[1] = ap[1];
        if (kc + 1 < KCN)
            __builtin_prefetch(Aswz + (((size_t)mtile*KCN + kc + 1)*32 + lane)*16, 0, 1);
#pragma unroll
        for (int i = 0; i < 4; ++i) {
            int nt = wave*4 + i;
            ABu b;
            const uint4* bp = (const uint4*)(Bswz + (((size_t)kc*NTN + nt)*32 + lane)*16);
            b.q[0] = bp[0];
            b.q[1] = bp[1];
            acc[i] = __builtin_amdgcn_wmma_f32_16x16x32_bf16(
                false, a.v, false, b.v, (short)0, acc[i], false, false);
        }
    }

    // C layout: lanes 0-15: VGPR v -> (M=v, N=lane); lanes 16-31: (M=v+8, N=lane-16)
    int rbase = (lane >= 16) ? 8 : 0;
#pragma unroll
    for (int i = 0; i < 4; ++i) {
        int nt = wave*4 + i;
        int col = nt*16 + (lane & 15);
        float wl  = Wl[(size_t)col*RED + col];  // diagonal of W_l
        float blc = bl[col];
#pragma unroll
        for (int vv = 0; vv < 8; ++vv) {
            int rl = vv + rbase;
            float eagg = 0.f;
            for (int rr = 0; rr < NR; ++rr) eagg += s_rels[rl*NR + rr] * s_redge[rr*RED + col];
            float g  = s_attn[rl]*acc[i][vv] + eagg;
            float hh = wl*g + blc;
            h[((size_t)(mtile*16 + rl))*RED + col] = hh > 0.f ? hh : 0.f;
        }
    }
}

// ---------------------------------------------------------------------------
// K3: per-batch reductions h_g / h_p over 72 (v,m) rows + FC + sigmoid.
// ---------------------------------------------------------------------------
__global__ void k3_final(const float* __restrict__ h, const float* __restrict__ fsum,
                         const float* __restrict__ fcW, const float* __restrict__ fcb,
                         float* __restrict__ out) {
    __shared__ float z[2*RED];
    int b = blockIdx.x;
    int r = threadIdx.x;                        // 256
    float hg = 0.f, hp = 0.f;
    for (int rl = 0; rl < NVIS*MF; ++rl) {
        int row = b*NVIS*MF + rl;
        float hv = h[(size_t)row*RED + r];
        hg += hv;
        hp += fsum[row]*hv;
    }
    z[r] = hg;
    z[RED + r] = hp;
    __syncthreads();
    if (threadIdx.x < OUTD) {
        int o = threadIdx.x;
        float acc = fcb[o];
        for (int j = 0; j < 2*RED; ++j) acc += z[j] * fcW[(size_t)o*2*RED + j];
        out[b*OUTD + o] = 1.f / (1.f + expf(-acc));
    }
}

// ---------------------------------------------------------------------------
// Workspace layout (bytes, 256-aligned blocks):
//   attn     @ 0          (2304*4   =     9216)
//   featsum  @ 9216       (2304*4   =     9216)
//   redge    @ 18432      (11*256*4 =    11264)
//   Bswz     @ 29696      (63*16*32*16*2 = 1032192)
//   Aswz     @ 1061888    (144*63*32*16*2 = 9289728)
//   h        @ 10351616   (2304*256*4 = 2359296)   total ~12.7 MB
// ---------------------------------------------------------------------------
extern "C" void kernel_launch(void* const* d_in, const int* in_sizes, int n_in,
                              void* d_out, int out_size, void* d_ws, size_t ws_size,
                              hipStream_t stream) {
    const float* feat   = (const float*)d_in[0];
    const float* neigh  = (const float*)d_in[1];
    const float* relidx = (const float*)d_in[2];
    const float* emb    = (const float*)d_in[3];
    const float* Wedge  = (const float*)d_in[4];
    const float* Wv     = (const float*)d_in[5];
    const float* Wr     = (const float*)d_in[6];
    const float* bv     = (const float*)d_in[7];
    const float* br     = (const float*)d_in[8];
    const float* Walpha = (const float*)d_in[9];
    const float* balpha = (const float*)d_in[10];
    const float* Wbeta  = (const float*)d_in[11];
    const float* bbeta  = (const float*)d_in[12];
    const float* Wl     = (const float*)d_in[13];
    const float* bl     = (const float*)d_in[14];
    const float* fcW    = (const float*)d_in[15];
    const float* fcb    = (const float*)d_in[16];

    char* ws = (char*)d_ws;
    float*          attn_ws  = (float*)(ws + 0);
    float*          fsum_ws  = (float*)(ws + 9216);
    float*          redge_ws = (float*)(ws + 18432);
    unsigned short* Bswz     = (unsigned short*)(ws + 29696);
    unsigned short* Aswz     = (unsigned short*)(ws + 1061888);
    float*          h_ws     = (float*)(ws + 10351616);
    float*          out      = (float*)d_out;

    k0_prep <<<KPAD + NR,   256, 0, stream>>>(emb, Wv, bv, Wedge, Wr, br, Bswz, redge_ws);
    k1_attn <<<BSZ*NVIS,    128, 0, stream>>>(feat, Walpha, balpha, Wbeta, bbeta, attn_ws, fsum_ws);
    k2a_tred<<<MTILES*KCN,  256, 0, stream>>>(neigh, Aswz);
    k2b_gemm<<<MTILES,      128, 0, stream>>>(Aswz, Bswz, relidx, redge_ws, attn_ws, Wl, bl, h_ws);
    k3_final<<<BSZ,         256, 0, stream>>>(h_ws, fsum_ws, fcW, fcb, out);
}
